// RegionModel_38371237822499
// MI455X (gfx1250) — matrix operand order repaired and checked by reading, not compiled, not run on hardware
//
#include <hip/hip_runtime.h>
#include <hip/hip_bf16.h>

typedef __attribute__((ext_vector_type(16))) __bf16       v16bf;
typedef __attribute__((ext_vector_type(8)))  float        v8f;
typedef __attribute__((ext_vector_type(4)))  unsigned int u32x4;

// ---------- helpers ----------
__device__ __forceinline__ unsigned short f2bf(float f) {
    unsigned int u = __float_as_uint(f);
    u += 0x7fffu + ((u >> 16) & 1u);       // round-to-nearest-even
    return (unsigned short)(u >> 16);
}

// ---------- kernel 0: convert expert weights fp32 -> bf16 ----------
__global__ void prep_weights(const float* __restrict__ w2, const float* __restrict__ w3,
                             unsigned short* __restrict__ w2bf, unsigned short* __restrict__ w3bf) {
    const int i  = blockIdx.x * 256 + threadIdx.x;
    const int n2 = 8 * 64 * 288;           // [R][64][32*9] row-major == [R][M][K]
    const int n3 = 8 * 128 * 576;          // [R][128][64*9]
    if (i < n2) w2bf[i] = f2bf(w2[i]);
    if (i < n3) w3bf[i] = f2bf(w3[i]);
}

// ---------- kernel 1: conv1 direct (3->32, s2p1) + ReLU, fp32 in, bf16 out ----------
__global__ void __launch_bounds__(256) conv1_direct(const float* __restrict__ image,
                                                    const float* __restrict__ w1,
                                                    const float* __restrict__ b1,
                                                    const int* __restrict__ region,
                                                    unsigned short* __restrict__ act1) {
    const int idx = blockIdx.x * 256 + threadIdx.x;    // [b][c][oy][ox], 128*32*128*128
    const int ox = idx & 127;
    const int oy = (idx >> 7) & 127;
    const int c  = (idx >> 14) & 31;
    const int b  = idx >> 19;
    const int r  = region[b];
    const float* img = image + (size_t)b * 3 * 65536;
    const float* w   = w1 + (r * 32 + c) * 27;
    float acc = b1[r * 32 + c];
    #pragma unroll
    for (int cin = 0; cin < 3; ++cin)
        #pragma unroll
        for (int ky = 0; ky < 3; ++ky) {
            const int iy = 2 * oy - 1 + ky;
            #pragma unroll
            for (int kx = 0; kx < 3; ++kx) {
                const int ix = 2 * ox - 1 + kx;
                if ((unsigned)iy < 256u && (unsigned)ix < 256u)
                    acc += img[cin * 65536 + (iy << 8) + ix] * w[cin * 9 + ky * 3 + kx];
            }
        }
    acc = acc > 0.f ? acc : 0.f;
    act1[idx] = f2bf(acc);
}

// ---------- kernel 2: conv2 implicit-GEMM WMMA (M=64, K=288, N=4096) ----------
// block = 1 sample x 8x8 output tile (64 cols); 8 waves x 2 wmma tiles; bf16 A/B, f32 acc
__global__ void __launch_bounds__(256) conv2_wmma(const unsigned short* __restrict__ act1,
                                                  const unsigned short* __restrict__ w2bf,
                                                  const float* __restrict__ b2,
                                                  const int* __restrict__ region,
                                                  unsigned short* __restrict__ act2) {
    __shared__ unsigned short sB[64 * 290];            // im2col tile, K padded 288->290 (odd dword stride)
    const int b   = blockIdx.y;
    const int s   = blockIdx.x;                        // 64 spatial tiles of 8x8 over 64x64
    const int r   = region[b];
    const int ty0 = (s >> 3) << 3, tx0 = (s & 7) << 3;
    const unsigned short* a1  = act1 + (size_t)b * 32 * 128 * 128;
    const unsigned short* w2r = w2bf + r * 64 * 288;

    // warm L2/L1 with this region's weights (global_prefetch_b8)
    for (int i = threadIdx.x; i < 576; i += 256)
        __builtin_prefetch(w2r + i * 32, 0, 3);

    // stage im2col: B[pos=0..63][k=0..287], k = c*9 + ky*3 + kx
    for (int e = threadIdx.x; e < 64 * 288; e += 256) {
        const int p = e / 288, k = e - p * 288;
        const int c = k / 9, rem = k - c * 9;
        const int ky = rem / 3, kx = rem - ky * 3;
        const int oy = ty0 + (p >> 3), ox = tx0 + (p & 7);
        const int iy = 2 * oy - 1 + ky, ix = 2 * ox - 1 + kx;
        unsigned short v = 0;
        if ((unsigned)iy < 128u && (unsigned)ix < 128u) v = a1[(c << 14) + (iy << 7) + ix];
        sB[p * 290 + k] = v;
    }
    __syncthreads();

    const int lane = threadIdx.x & 31, wave = threadIdx.x >> 5;
    const int half = lane >> 4, l16 = lane & 15;
    const int mtile = wave >> 1;                       // 4 mtiles (64 channels)
    const int n0    = (wave & 1) * 2;                  // 2 of 4 ntiles per wave
    v8f acc0 = {}; v8f acc1 = {};
    const unsigned short* arow = w2r + (mtile * 16 + l16) * 288;   // A: lane = row M

    for (int k0 = 0; k0 < 288; k0 += 32) {
        union { v16bf v; u32x4 q[2]; } A;
        const int ka = k0 + half * 16;                 // A layout: half selects K 0..15 / 16..31 (contiguous)
        A.q[0] = *(const u32x4*)(arow + ka);
        A.q[1] = *(const u32x4*)(arow + ka + 8);
        union { v16bf v; unsigned int u[8]; } Bf0, Bf1;
        const int kb = k0 + half * 16;                 // B layout: lane = col N, 16 consecutive K
        const unsigned int* r0 = (const unsigned int*)(sB + ((n0    ) * 16 + l16) * 290 + kb);
        const unsigned int* r1 = (const unsigned int*)(sB + ((n0 + 1) * 16 + l16) * 290 + kb);
        #pragma unroll
        for (int j = 0; j < 8; ++j) { Bf0.u[j] = r0[j]; Bf1.u[j] = r1[j]; }
        acc0 = __builtin_amdgcn_wmma_f32_16x16x32_bf16(false, A.v, false, Bf0.v, (short)0, acc0, false, false);
        acc1 = __builtin_amdgcn_wmma_f32_16x16x32_bf16(false, A.v, false, Bf1.v, (short)0, acc1, false, false);
    }

    // epilogue: bias + ReLU, store bf16 act2 [b][64][64][64]
    unsigned short* o = act2 + (size_t)b * 64 * 64 * 64;
    #pragma unroll
    for (int i = 0; i < 8; ++i) {
        const int c = mtile * 16 + half * 8 + i;       // C/D layout: lanes 16-31 hold M+8
        const float bias = b2[r * 64 + c];
        float v0 = acc0[i] + bias; v0 = v0 > 0.f ? v0 : 0.f;
        float v1 = acc1[i] + bias; v1 = v1 > 0.f ? v1 : 0.f;
        const int np0 = n0 * 16 + l16, np1 = np0 + 16;
        o[(c << 12) + ((ty0 + (np0 >> 3)) << 6) + (tx0 + (np0 & 7))] = f2bf(v0);
        o[(c << 12) + ((ty0 + (np1 >> 3)) << 6) + (tx0 + (np1 & 7))] = f2bf(v1);
    }
}

// ---------- kernel 3: conv3 WMMA (M=128, K=576, N=1024) fused with partial avg-pool ----------
__global__ void __launch_bounds__(256) conv3_wmma(const unsigned short* __restrict__ act2,
                                                  const unsigned short* __restrict__ w3bf,
                                                  const float* __restrict__ b3,
                                                  const int* __restrict__ region,
                                                  float* __restrict__ partial) {
    __shared__ unsigned short sB[32 * 578];            // 32 positions x K(576->578 padded)
    const int b   = blockIdx.y;
    const int s   = blockIdx.x;                        // 32 tiles: 4 rows x 8 cols over 32x32
    const int r   = region[b];
    const int ty0 = (s >> 2) << 2, tx0 = (s & 3) << 3;
    const unsigned short* a2  = act2 + (size_t)b * 64 * 64 * 64;
    const unsigned short* w3r = w3bf + r * 128 * 576;

    for (int i = threadIdx.x; i < 2304; i += 256)      // prefetch 128*576*2B = 144KB region
        __builtin_prefetch(w3r + i * 32, 0, 3);

    for (int e = threadIdx.x; e < 32 * 576; e += 256) {
        const int p = e / 576, k = e - p * 576;
        const int c = k / 9, rem = k - c * 9;
        const int ky = rem / 3, kx = rem - ky * 3;
        const int oy = ty0 + (p >> 3), ox = tx0 + (p & 7);
        const int iy = 2 * oy - 1 + ky, ix = 2 * ox - 1 + kx;
        unsigned short v = 0;
        if ((unsigned)iy < 64u && (unsigned)ix < 64u) v = a2[(c << 12) + (iy << 6) + ix];
        sB[p * 578 + k] = v;
    }
    __syncthreads();

    const int lane = threadIdx.x & 31, wave = threadIdx.x >> 5;
    const int half = lane >> 4, l16 = lane & 15;
    const int mtile = wave;                            // 8 mtiles = 128 channels
    v8f acc0 = {}; v8f acc1 = {};
    const unsigned short* arow = w3r + (mtile * 16 + l16) * 576;

    for (int k0 = 0; k0 < 576; k0 += 32) {
        union { v16bf v; u32x4 q[2]; } A;
        const int ka = k0 + half * 16;
        A.q[0] = *(const u32x4*)(arow + ka);
        A.q[1] = *(const u32x4*)(arow + ka + 8);
        union { v16bf v; unsigned int u[8]; } Bf0, Bf1;
        const int kb = k0 + half * 16;
        const unsigned int* r0 = (const unsigned int*)(sB + (l16     ) * 578 + kb);
        const unsigned int* r1 = (const unsigned int*)(sB + (l16 + 16) * 578 + kb);
        #pragma unroll
        for (int j = 0; j < 8; ++j) { Bf0.u[j] = r0[j]; Bf1.u[j] = r1[j]; }
        acc0 = __builtin_amdgcn_wmma_f32_16x16x32_bf16(false, A.v, false, Bf0.v, (short)0, acc0, false, false);
        acc1 = __builtin_amdgcn_wmma_f32_16x16x32_bf16(false, A.v, false, Bf1.v, (short)0, acc1, false, false);
    }

    // bias + ReLU + sum over this block's 32 positions, deterministic partial per (b, s, c)
    #pragma unroll
    for (int i = 0; i < 8; ++i) {
        const int c = mtile * 16 + half * 8 + i;
        const float bias = b3[r * 128 + c];
        float v0 = acc0[i] + bias; v0 = v0 > 0.f ? v0 : 0.f;
        float v1 = acc1[i] + bias; v1 = v1 > 0.f ? v1 : 0.f;
        float v = v0 + v1;
        #pragma unroll
        for (int m = 1; m <= 8; m <<= 1) v += __shfl_xor(v, m, 32);   // reduce 16-lane half
        if (l16 == 0) partial[((b * 32 + s) * 128) + c] = v;
    }
}

// ---------- kernel 4: reduce partials -> pooled features ----------
__global__ void pool_feat(const float* __restrict__ partial, float* __restrict__ feat) {
    const int b = blockIdx.x, c = threadIdx.x;         // 128 x 128
    float sum = 0.f;
    for (int s = 0; s < 32; ++s) sum += partial[(b * 32 + s) * 128 + c];
    feat[b * 128 + c] = sum * (1.f / 1024.f);
}

// ---------- kernel 5: per-region FC head ----------
__global__ void fc_head(const float* __restrict__ feat, const float* __restrict__ fw,
                        const float* __restrict__ fb, const int* __restrict__ region,
                        float* __restrict__ out) {
    const int t = blockIdx.x * 256 + threadIdx.x;
    if (t >= 256) return;
    const int b = t >> 1, cls = t & 1;
    const int r = region[b];
    const float* w = fw + (r * 2 + cls) * 128;
    const float* f = feat + b * 128;
    float sum = fb[r * 2 + cls];
    for (int c = 0; c < 128; ++c) sum += w[c] * f[c];
    out[t] = sum;
}

// ---------- launch ----------
extern "C" void kernel_launch(void* const* d_in, const int* in_sizes, int n_in,
                              void* d_out, int out_size, void* d_ws, size_t ws_size,
                              hipStream_t stream) {
    (void)in_sizes; (void)n_in; (void)out_size; (void)ws_size;
    const float* image  = (const float*)d_in[0];
    const int*   region = (const int*)  d_in[1];
    const float* w1     = (const float*)d_in[2];
    const float* b1     = (const float*)d_in[3];
    const float* w2     = (const float*)d_in[4];
    const float* b2     = (const float*)d_in[5];
    const float* w3     = (const float*)d_in[6];
    const float* b3     = (const float*)d_in[7];
    const float* fw     = (const float*)d_in[8];
    const float* fb     = (const float*)d_in[9];
    float* out = (float*)d_out;

    char* ws = (char*)d_ws;
    size_t off = 0;
    auto carve = [&](size_t bytes) { void* p = ws + off; off = (off + bytes + 255) & ~(size_t)255; return p; };
    unsigned short* w2bf    = (unsigned short*)carve((size_t)8 * 64 * 288 * 2);        // 288 KB
    unsigned short* w3bf    = (unsigned short*)carve((size_t)8 * 128 * 576 * 2);       // 1.1 MB
    unsigned short* act1    = (unsigned short*)carve((size_t)128 * 32 * 128 * 128 * 2);// 134 MB bf16
    unsigned short* act2    = (unsigned short*)carve((size_t)128 * 64 * 64 * 64 * 2);  // 67 MB bf16
    float*          partial = (float*)carve((size_t)128 * 32 * 128 * 4);               // 2 MB
    float*          feat    = (float*)carve((size_t)128 * 128 * 4);                    // 64 KB

    prep_weights<<<(8 * 128 * 576 + 255) / 256, 256, 0, stream>>>(w2, w3, w2bf, w3bf);
    conv1_direct<<<(128 * 32 * 128 * 128) / 256, 256, 0, stream>>>(image, w1, b1, region, act1);
    conv2_wmma<<<dim3(64, 128), 256, 0, stream>>>(act1, w2bf, b2, region, act2);
    conv3_wmma<<<dim3(32, 128), 256, 0, stream>>>(act2, w3bf, b3, region, partial);
    pool_feat<<<128, 128, 0, stream>>>(partial, feat);
    fc_head<<<1, 256, 0, stream>>>(feat, fw, fb, region, out);
}
